// GNNClassifier_88648124990772
// MI455X (gfx1250) — compile-verified, matrix operand order checked by reading, not run on hardware
//
#include <hip/hip_runtime.h>
#include <hip/hip_bf16.h>

#define HID 64

typedef __attribute__((ext_vector_type(16))) __bf16 v16bf;
typedef __attribute__((ext_vector_type(8)))  float  v8f;

static __device__ inline __bf16 f2bf(float f) {
    unsigned u = __float_as_uint(f);
    unsigned r = (u + 0x7FFFu + ((u >> 16) & 1u)) >> 16;   // round-to-nearest-even
    unsigned short s = (unsigned short)r;
    __bf16 b;
    __builtin_memcpy(&b, &s, 2);
    return b;
}

// ---------------------------------------------------------------------------
// Zero helper (workspace is poisoned; zero accumulators each launch)
// ---------------------------------------------------------------------------
__global__ void k_zero(float* __restrict__ p, long n) {
    long t = (long)blockIdx.x * blockDim.x + threadIdx.x;
    long stride = (long)gridDim.x * blockDim.x;
    for (long i = t; i < n; i += stride) p[i] = 0.0f;
}

// ---------------------------------------------------------------------------
// Fold embeddings through W_in:  Ts = shape_emb @ W_in[:64,:], Tc = color @ W_in[64:,:]
// ---------------------------------------------------------------------------
__global__ void k_tables(const float* __restrict__ shape_emb,
                         const float* __restrict__ color_emb,
                         const float* __restrict__ W_in,
                         float* __restrict__ Ts, float* __restrict__ Tc) {
    int t = blockIdx.x * blockDim.x + threadIdx.x;
    if (t < 16 * HID) {
        int i = t >> 6, j = t & 63;
        float s = 0.0f;
        for (int k = 0; k < HID; ++k) s += shape_emb[i * HID + k] * W_in[k * HID + j];
        Ts[t] = s;
    } else if (t < (16 + 8) * HID) {
        int t2 = t - 16 * HID;
        int i = t2 >> 6, j = t2 & 63;
        float s = 0.0f;
        for (int k = 0; k < HID; ++k) s += color_emb[i * HID + k] * W_in[(HID + k) * HID + j];
        Tc[t2] = s;
    }
}

// h0[n] = relu(Ts[x0] + Tc[x1] + b_in) ; thread handles 4 channels
__global__ void k_embed(const int* __restrict__ x,
                        const float* __restrict__ Ts, const float* __restrict__ Tc,
                        const float* __restrict__ b_in,
                        float* __restrict__ h0, int N) {
    int t = blockIdx.x * blockDim.x + threadIdx.x;
    if (t >= N * 16) return;
    int n = t >> 4, cg = (t & 15) * 4;
    int s = x[n * 2 + 0], c = x[n * 2 + 1];
    float4 a  = *(const float4*)&Ts[s * HID + cg];
    float4 b  = *(const float4*)&Tc[c * HID + cg];
    float4 bi = *(const float4*)&b_in[cg];
    float4 o;
    o.x = fmaxf(a.x + b.x + bi.x, 0.0f);
    o.y = fmaxf(a.y + b.y + bi.y, 0.0f);
    o.z = fmaxf(a.z + b.z + bi.z, 0.0f);
    o.w = fmaxf(a.w + b.w + bi.w, 0.0f);
    *(float4*)&h0[(size_t)n * HID + cg] = o;
}

// ---------------------------------------------------------------------------
// Degree / graph-count histograms
// ---------------------------------------------------------------------------
__global__ void k_deg(const int* __restrict__ ei, float* __restrict__ deg, int E) {
    int e = blockIdx.x * blockDim.x + threadIdx.x;
    if (e < E) atomicAdd(&deg[ei[E + e]], 1.0f);   // dst row
}
__global__ void k_cnt(const int* __restrict__ batch, float* __restrict__ cnt, int N) {
    int n = blockIdx.x * blockDim.x + threadIdx.x;
    if (n < N) atomicAdd(&cnt[batch[n]], 1.0f);
}

// agg[dst] += h[src] ; thread = (edge, 4-channel group)
__global__ void k_scatter(const int* __restrict__ ei, const float* __restrict__ h,
                          float* __restrict__ agg, int E) {
    int t = blockIdx.x * blockDim.x + threadIdx.x;
    if (t >= E * 16) return;
    int e = t >> 4, cg = (t & 15) * 4;
    int src = ei[e], dst = ei[E + e];
    float4 v = *(const float4*)&h[(size_t)src * HID + cg];
    float* a = &agg[(size_t)dst * HID + cg];
    atomicAdd(a + 0, v.x);
    atomicAdd(a + 1, v.y);
    atomicAdd(a + 2, v.z);
    atomicAdd(a + 3, v.w);
}

// pooled[batch[n]] += h[n]
__global__ void k_pool(const int* __restrict__ batch, const float* __restrict__ h,
                       float* __restrict__ pooled, int N) {
    int t = blockIdx.x * blockDim.x + threadIdx.x;
    if (t >= N * 16) return;
    int n = t >> 4, cg = (t & 15) * 4;
    float4 v = *(const float4*)&h[(size_t)n * HID + cg];
    float* p = &pooled[(size_t)batch[n] * HID + cg];
    atomicAdd(p + 0, v.x);
    atomicAdd(p + 1, v.y);
    atomicAdd(p + 2, v.z);
    atomicAdd(p + 3, v.w);
}

// ---------------------------------------------------------------------------
// Fused SAGE dense step on WMMA:
//   h_out = relu( (agg/max(deg,1)) @ Wl + h_in @ Wr + b )
// One wave computes a 16-row x 64-col output block.
// bf16 A/B fragments, f32 accumulation via v_wmma_f32_16x16x32_bf16.
// ---------------------------------------------------------------------------
__global__ __launch_bounds__(256)
void k_sage_wmma(const float* __restrict__ h_in,   // [N,64]
                 const float* __restrict__ agg,    // [N,64] unnormalized sums
                 const float* __restrict__ deg,    // [N]
                 const float* __restrict__ Wl,     // [64,64] row-major (k,n)
                 const float* __restrict__ Wr,     // [64,64]
                 const float* __restrict__ bias,   // [64]
                 float* __restrict__ h_out,        // [N,64]
                 int N) {
    // Weights in LDS, column-major (index n*64+k) so B fragments are contiguous.
    __shared__ __bf16 sWl[HID * HID];
    __shared__ __bf16 sWr[HID * HID];
    for (int i = threadIdx.x; i < HID * HID; i += blockDim.x) {
        int k = i & 63, n = i >> 6;
        sWl[i] = f2bf(Wl[k * HID + n]);
        sWr[i] = f2bf(Wr[k * HID + n]);
    }
    __syncthreads();

    const int wave = threadIdx.x >> 5;
    const int lane = threadIdx.x & 31;
    const int m0 = (blockIdx.x * 8 + wave) * 16;
    if (m0 >= N) return;                        // wave-uniform exit
    const bool full = (m0 + 16 <= N);           // wave-uniform fast path

    const int rlo = lane & 15;
    int r = m0 + rlo;
    if (r >= N) r = N - 1;                      // clamp pad rows (stores guarded)
    const int kb = (lane < 16) ? 0 : 8;         // 16-bit A-matrix K sub-layout

    const float dv  = deg[r];
    const float inv = 1.0f / fmaxf(dv, 1.0f);

    // A fragments: Aa = normalized aggregate rows, Ah = self rows. 2 K-steps.
    v16bf Aa[2], Ah[2];
    const float* hrow = h_in + (size_t)r * HID;
    const float* arow = agg  + (size_t)r * HID;
#pragma unroll
    for (int ks = 0; ks < 2; ++ks) {
        const int k0 = ks * 32 + kb;
#pragma unroll
        for (int j = 0; j < 8; ++j) {
            Aa[ks][j]     = f2bf(arow[k0 + j] * inv);
            Aa[ks][j + 8] = f2bf(arow[k0 + 16 + j] * inv);
            Ah[ks][j]     = f2bf(hrow[k0 + j]);
            Ah[ks][j + 8] = f2bf(hrow[k0 + 16 + j]);
        }
    }

    const int rbase = m0 + ((lane >= 16) ? 8 : 0);   // f32 C/D row mapping
#pragma unroll
    for (int nt = 0; nt < 4; ++nt) {
        const int n0 = nt * 16;
        const int c  = n0 + rlo;                     // this lane's output column
        const __bf16* bl = &sWl[c * HID];
        const __bf16* br = &sWr[c * HID];
        v16bf Bl[2], Br[2];
#pragma unroll
        for (int ks = 0; ks < 2; ++ks) {
            const int k0 = ks * 32 + kb;
#pragma unroll
            for (int j = 0; j < 8; ++j) {
                Bl[ks][j]     = bl[k0 + j];
                Bl[ks][j + 8] = bl[k0 + 16 + j];
                Br[ks][j]     = br[k0 + j];
                Br[ks][j + 8] = br[k0 + 16 + j];
            }
        }
        const float bb = bias[c];
        v8f acc = {bb, bb, bb, bb, bb, bb, bb, bb};
        acc = __builtin_amdgcn_wmma_f32_16x16x32_bf16(false, Aa[0], false, Bl[0], (short)0, acc, false, false);
        acc = __builtin_amdgcn_wmma_f32_16x16x32_bf16(false, Aa[1], false, Bl[1], (short)0, acc, false, false);
        acc = __builtin_amdgcn_wmma_f32_16x16x32_bf16(false, Ah[0], false, Br[0], (short)0, acc, false, false);
        acc = __builtin_amdgcn_wmma_f32_16x16x32_bf16(false, Ah[1], false, Br[1], (short)0, acc, false, false);

        // Epilogue: relu + store. One base address per lane; unrolled row
        // offsets become immediate instruction offsets (coalesced across lanes).
        float* p = h_out + (size_t)rbase * HID + c;
        if (full) {
#pragma unroll
            for (int j = 0; j < 8; ++j) {
                const float v = acc[j];
                p[(size_t)j * HID] = v > 0.0f ? v : 0.0f;
            }
        } else {
#pragma unroll
            for (int j = 0; j < 8; ++j) {
                if (rbase + j < N) {
                    const float v = acc[j];
                    p[(size_t)j * HID] = v > 0.0f ? v : 0.0f;
                }
            }
        }
    }
}

// ---------------------------------------------------------------------------
// Head: out[g,c] = b_out[c] + sum_k pooled[g,k]/max(cnt[g],1) * W_out[k,c]
// ---------------------------------------------------------------------------
__global__ void k_head(const float* __restrict__ pooled, const float* __restrict__ cnt,
                       const float* __restrict__ W_out, const float* __restrict__ b_out,
                       float* __restrict__ out, int G) {
    int t = blockIdx.x * blockDim.x + threadIdx.x;
    if (t >= G * 2) return;
    int g = t >> 1, c = t & 1;
    float inv = 1.0f / fmaxf(cnt[g], 1.0f);
    float s = 0.0f;
    for (int k = 0; k < HID; ++k) s += pooled[g * HID + k] * W_out[k * 2 + c];
    out[t] = s * inv + b_out[c];
}

extern "C" void kernel_launch(void* const* d_in, const int* in_sizes, int n_in,
                              void* d_out, int out_size, void* d_ws, size_t ws_size,
                              hipStream_t stream) {
    const int*   x         = (const int*)  d_in[0];
    const int*   ei        = (const int*)  d_in[1];
    const int*   batch     = (const int*)  d_in[2];
    const float* shape_emb = (const float*)d_in[4];
    const float* color_emb = (const float*)d_in[5];
    const float* W_in      = (const float*)d_in[6];
    const float* b_in      = (const float*)d_in[7];
    const float* W1l       = (const float*)d_in[8];
    const float* b1        = (const float*)d_in[9];
    const float* W1r       = (const float*)d_in[10];
    const float* W2l       = (const float*)d_in[11];
    const float* b2        = (const float*)d_in[12];
    const float* W2r       = (const float*)d_in[13];
    const float* W_out     = (const float*)d_in[14];
    const float* b_out     = (const float*)d_in[15];

    const int N = in_sizes[0] / 2;         // nodes
    const int E = in_sizes[1] / 2;         // edges
    const int G = 512;                     // N_GRAPHS (compile-time constant in reference)

    // Workspace carve-up (all f32)
    float* h0     = (float*)d_ws;                     // N*64
    float* h1     = h0 + (size_t)N * HID;             // N*64
    float* agg    = h1 + (size_t)N * HID;             // N*64
    float* deg    = agg + (size_t)N * HID;            // N
    float* pooled = deg + N;                          // G*64
    float* cnt    = pooled + (size_t)G * HID;         // G
    float* Ts     = cnt + G;                          // 16*64
    float* Tc     = Ts + 16 * HID;                    // 8*64

    float* out = (float*)d_out;

    // 1. fold embeddings through W_in
    k_tables<<<6, 256, 0, stream>>>(shape_emb, color_emb, W_in, Ts, Tc);

    // 2. zero deg+pooled+cnt (contiguous) and agg
    k_zero<<<1024, 256, 0, stream>>>(deg, (long)N + (long)G * HID + G);
    k_zero<<<2048, 256, 0, stream>>>(agg, (long)N * HID);

    // 3. node features h0
    k_embed<<<(N * 16 + 255) / 256, 256, 0, stream>>>(x, Ts, Tc, b_in, h0, N);

    // 4. degrees & graph counts
    k_deg<<<(E + 255) / 256, 256, 0, stream>>>(ei, deg, E);
    k_cnt<<<(N + 255) / 256, 256, 0, stream>>>(batch, cnt, N);

    const int sageBlocks = (N + 127) / 128;   // 8 waves x 16 rows per block

    // 5. SAGE conv 1: h0 -> h1
    k_scatter<<<(E * 16 + 255) / 256, 256, 0, stream>>>(ei, h0, agg, E);
    k_sage_wmma<<<sageBlocks, 256, 0, stream>>>(h0, agg, deg, W1l, W1r, b1, h1, N);

    // 6. SAGE conv 2: h1 -> h0 (reuse buffer)
    k_zero<<<2048, 256, 0, stream>>>(agg, (long)N * HID);
    k_scatter<<<(E * 16 + 255) / 256, 256, 0, stream>>>(ei, h1, agg, E);
    k_sage_wmma<<<sageBlocks, 256, 0, stream>>>(h1, agg, deg, W2l, W2r, b2, h0, N);

    // 7. mean pool + classifier head
    k_pool<<<(N * 16 + 255) / 256, 256, 0, stream>>>(batch, h0, pooled, N);
    k_head<<<(G * 2 + 255) / 256, 256, 0, stream>>>(pooled, cnt, W_out, b_out, out, G);
}
